// BiLSTM_CRF_64192581206655
// MI455X (gfx1250) — matrix operand order repaired and checked by reading, not compiled, not run on hardware
//
#include <hip/hip_runtime.h>
#include <hip/hip_bf16.h>
#include <stdint.h>
#include <stddef.h>

// ---------------- problem constants ----------------
#define T_LEN   4096
#define EMB     1024
#define H2      512
#define RGATE   2048        // 4*H2 gate rows per direction
#define NTAGS   5
#define START_TAG 3
#define STOP_TAG  4
#define NEG_INF (-10000.0f)

// recurrence sharding
#define NWG_DIR  32         // workgroups per direction
#define UNITS_WG 16         // hidden units owned per WG  (H2/NWG_DIR)
#define ROWS_WG  64         // gate rows per WG (4*UNITS_WG)

// GEMM blocking
#define BM 64               // block tile M (time)
#define BN 128              // block tile N (gate rows)
#define BK 32               // K panel depth (one WMMA K-step)
#define LPAD 40             // LDS row stride in ushorts (80B: 16B-aligned, conflict-free)

typedef __attribute__((ext_vector_type(16))) __bf16          v16bf;
typedef __attribute__((ext_vector_type(8)))  float           v8f;
typedef __attribute__((ext_vector_type(8)))  unsigned short  v8us;
typedef __attribute__((ext_vector_type(16))) unsigned short  v16us;
typedef __attribute__((ext_vector_type(4)))  int             v4i;

__device__ __forceinline__ unsigned short f2bf(float f) {
  // round-to-nearest-even f32 -> bf16
  unsigned u = __float_as_uint(f);
  u += 0x7FFFu + ((u >> 16) & 1u);
  return (unsigned short)(u >> 16);
}

// ---- CDNA5 async global->LDS staging (ASYNCcnt path), with safe fallback ----
__device__ __forceinline__ void async_ld16(const unsigned short* g, unsigned short* l) {
#if __has_builtin(__builtin_amdgcn_global_load_async_to_lds_b128)
  typedef __attribute__((address_space(1))) v4i* gp_t;   // global (AS1), int4 payload
  typedef __attribute__((address_space(3))) v4i* lp_t;   // LDS (AS3), int4 payload
  __builtin_amdgcn_global_load_async_to_lds_b128((gp_t)g, (lp_t)l, 0, 0);
#else
  *(v8us*)l = *(const v8us*)g;     // synchronous fallback
#endif
}

__device__ __forceinline__ void wait_async0() {
#if __has_builtin(__builtin_amdgcn_s_wait_asynccnt)
  __builtin_amdgcn_s_wait_asynccnt(0);
#else
  asm volatile("s_wait_asynccnt 0x0" ::: "memory");
#endif
}

__device__ __forceinline__ v16bf ld_frag(const unsigned short* p0, const unsigned short* p1) {
  v8us lo = *(const v8us*)p0;
  v8us hi = *(const v8us*)p1;
  v16us u = __builtin_shufflevector(lo, hi, 0,1,2,3,4,5,6,7,8,9,10,11,12,13,14,15);
  return __builtin_bit_cast(v16bf, u);
}

// ---------------------------------------------------------------------------
// Kernel 0a: embedding gather + f32->bf16 pack, row-major [T, EMB]
// ---------------------------------------------------------------------------
__global__ void __launch_bounds__(256)
k_gather_bf16(const int* __restrict__ sentence,
              const float* __restrict__ embed,
              unsigned short* __restrict__ Xbf)
{
  int t = blockIdx.x;
  const float* src = embed + (size_t)sentence[t] * EMB;
  unsigned short* dst = Xbf + (size_t)t * EMB;
  for (int k = threadIdx.x; k < EMB; k += 256)
    dst[k] = f2bf(src[k]);
}

// ---------------------------------------------------------------------------
// Kernel 0b: generic f32->bf16 convert (for Wih_f / Wih_b)
// ---------------------------------------------------------------------------
__global__ void __launch_bounds__(256)
k_cvt_bf16(const float* __restrict__ src, unsigned short* __restrict__ dst, int n)
{
  int i = blockIdx.x * 256 + threadIdx.x;
  if (i < n) dst[i] = f2bf(src[i]);
}

// ---------------------------------------------------------------------------
// Kernel 0c: bias sum  bsum = bih + bhh
// ---------------------------------------------------------------------------
__global__ void __launch_bounds__(256)
k_bias_sum(const float* __restrict__ a, const float* __restrict__ b,
           float* __restrict__ o, int n)
{
  int i = blockIdx.x * 256 + threadIdx.x;
  if (i < n) o[i] = a[i] + b[i];
}

// ---------------------------------------------------------------------------
// Kernel 0d: init h double-buffers from h0, zero the grid-sync counters.
// ---------------------------------------------------------------------------
__global__ void __launch_bounds__(256)
k_init(const float* __restrict__ h0, float* __restrict__ hbuf,
       unsigned* __restrict__ sync)
{
  int i = blockIdx.x * 256 + threadIdx.x;
  if (i < 4) sync[i] = 0u;
  if (i < 2 * H2) {
    int dir = i >> 9, u = i & (H2 - 1);
    hbuf[dir * (2 * H2) + u] = h0[i];          // buffer 0 = initial h
  }
}

// ---------------------------------------------------------------------------
// Kernel 1: bf16 WMMA GEMM  G[t,r] = sum_k X[t,k]*Wih[r,k] + (bih[r]+bhh[r])
// Block = 64x128 tile, 8 waves (2x4), each wave 32x32 (four 16x16 accums).
// A/B panels async-staged into double-buffered LDS (ASYNCcnt), fragments read
// from LDS per the ISA WMMA layouts (CDNA5 ISA §7.12.2).
// ---------------------------------------------------------------------------
__global__ void __launch_bounds__(256)
k_gemm_wmma(const unsigned short* __restrict__ X,
            const unsigned short* __restrict__ Wf,
            const unsigned short* __restrict__ Wb,
            const float* __restrict__ bsf, const float* __restrict__ bsb,
            float* __restrict__ Gf, float* __restrict__ Gb)
{
  __shared__ unsigned short Al[2][BM][LPAD];   // 2*64*40*2  = 10.0 KB
  __shared__ unsigned short Bl[2][BN][LPAD];   // 2*128*40*2 = 20.0 KB

  const int tid  = threadIdx.x;
  const int lane = tid & 31;
  const int wave = tid >> 5;
  const int tBase = blockIdx.x * BM;
  const int rBase = blockIdx.y * BN;
  const int dir   = blockIdx.z;

  const unsigned short* W  = dir ? Wb  : Wf;
  const float*          bs = dir ? bsb : bsf;
  float*                G  = dir ? Gb  : Gf;

  // staging role: each thread moves one 16B chunk of A and two of B per panel
  const int srow = tid >> 2;             // 0..63
  const int scol = (tid & 3) * 8;        // 0,8,16,24 (bf16 elements)
  const unsigned short* gA  = X + (size_t)(tBase + srow) * EMB + scol;
  const unsigned short* gB0 = W + (size_t)(rBase + srow) * EMB + scol;
  const unsigned short* gB1 = W + (size_t)(rBase + 64 + srow) * EMB + scol;

  // compute role
  const int wm = (wave >> 2) * 32;       // wave M offset in block tile
  const int wn = (wave & 3) * 32;        // wave N offset
  const int m   = lane & 15;
  const int sub = lane >> 4;

  v8f acc[2][2] = {};

  // prologue: stage panel 0
  async_ld16(gA,  &Al[0][srow][scol]);
  async_ld16(gB0, &Bl[0][srow][scol]);
  async_ld16(gB1, &Bl[0][64 + srow][scol]);
  wait_async0();
  __syncthreads();

  for (int kk = 0; kk < EMB / BK; ++kk) {
    const int buf = kk & 1;
    if (kk + 1 < EMB / BK) {             // stage next panel while computing
      const int ko = (kk + 1) * BK;
      async_ld16(gA  + ko, &Al[buf ^ 1][srow][scol]);
      async_ld16(gB0 + ko, &Bl[buf ^ 1][srow][scol]);
      async_ld16(gB1 + ko, &Bl[buf ^ 1][64 + srow][scol]);
    }

    // A fragments: lane sub=0 holds K {0..7,16..23}, sub=1 holds {8..15,24..31}
    v16bf Afr[2], Bfr[2];
    #pragma unroll
    for (int mt = 0; mt < 2; ++mt) {
      const unsigned short* ar = &Al[buf][wm + mt * 16 + m][0];
      Afr[mt] = ld_frag(ar + sub * 8, ar + 16 + sub * 8);
    }
    // B fragments: lane holds column n, K {0..7,16..23} for sub=0, {8..15,24..31} for sub=1
    #pragma unroll
    for (int nt = 0; nt < 2; ++nt) {
      const unsigned short* br = &Bl[buf][wn + nt * 16 + m][sub * 8];
      Bfr[nt] = ld_frag(br, br + 16);
    }
    #pragma unroll
    for (int mt = 0; mt < 2; ++mt)
      #pragma unroll
      for (int nt = 0; nt < 2; ++nt)
        acc[mt][nt] = __builtin_amdgcn_wmma_f32_16x16x32_bf16(
            false, Afr[mt], false, Bfr[nt], (short)0, acc[mt][nt], false, false);

    wait_async0();
    __syncthreads();
  }

  // epilogue: add bias, store f32
  #pragma unroll
  for (int nt = 0; nt < 2; ++nt) {
    const int col  = rBase + wn + nt * 16 + m;
    const float bb = bs[col];
    #pragma unroll
    for (int mt = 0; mt < 2; ++mt) {
      const int row0 = tBase + wm + mt * 16 + sub * 8;
      #pragma unroll
      for (int v = 0; v < 8; ++v)
        G[(size_t)(row0 + v) * RGATE + col] = acc[mt][nt][v] + bb;
    }
  }
}

// ---------------------------------------------------------------------------
// Kernel 2: persistent bidirectional LSTM recurrence.
// 64 workgroups: blocks 0..31 forward, 32..63 backward. Each WG owns 16 hidden
// units; its 64 gate rows of Whh live in dynamic LDS (128 KB, f32). h is
// double-buffered in global; one device-scope sense-reversing barrier / step.
// ---------------------------------------------------------------------------
__device__ __forceinline__ void grid_sync(unsigned* cnt, unsigned* phase)
{
  __syncthreads();
  if (threadIdx.x == 0) {
    __threadfence();
    unsigned ph = __hip_atomic_load(phase, __ATOMIC_ACQUIRE, __HIP_MEMORY_SCOPE_AGENT);
    unsigned arrived = __hip_atomic_fetch_add(cnt, 1u, __ATOMIC_ACQ_REL,
                                              __HIP_MEMORY_SCOPE_AGENT);
    if (arrived == NWG_DIR - 1) {
      __hip_atomic_store(cnt, 0u, __ATOMIC_RELAXED, __HIP_MEMORY_SCOPE_AGENT);
      __hip_atomic_fetch_add(phase, 1u, __ATOMIC_RELEASE, __HIP_MEMORY_SCOPE_AGENT);
    } else {
      while (__hip_atomic_load(phase, __ATOMIC_ACQUIRE, __HIP_MEMORY_SCOPE_AGENT) == ph)
        __builtin_amdgcn_s_sleep(1);
    }
  }
  __syncthreads();
}

__global__ void __launch_bounds__(256)
k_lstm_rec(const float* __restrict__ Whh_f, const float* __restrict__ Whh_b,
           const float* __restrict__ c0,
           const float* __restrict__ Gf, const float* __restrict__ Gb,
           float* __restrict__ hbuf,     // [2][2][H2]
           float* __restrict__ hsf, float* __restrict__ hsb,   // [T][H2]
           unsigned* __restrict__ sync)  // [4]: cnt_f,cnt_b,ph_f,ph_b
{
  extern __shared__ float lds[];
  float* Wl      = lds;                       // [ROWS_WG * H2]
  float* hprev   = Wl + ROWS_WG * H2;         // [H2]
  float* partial = hprev + H2;                // [256]
  float* gate    = partial + 256;             // [ROWS_WG]

  const int wg  = blockIdx.x & (NWG_DIR - 1);
  const int dir = blockIdx.x >> 5;
  const float* Whh = dir ? Whh_b : Whh_f;
  const float* G   = dir ? Gb : Gf;
  float*       hs  = dir ? hsb : hsf;
  unsigned* cnt = sync + dir;
  unsigned* phs = sync + 2 + dir;
  const int u0  = wg * UNITS_WG;
  const int tid = threadIdx.x;

  // stage this WG's Whh slice: local row lr = q*16+j  <->  global row q*H2+u0+j
  for (int i = tid; i < ROWS_WG * H2; i += 256) {
    int lr = i >> 9, k = i & (H2 - 1);
    int q = lr >> 4, j = lr & 15;
    Wl[i] = Whh[(size_t)(q * H2 + u0 + j) * H2 + k];
  }
  float creg = (tid < UNITS_WG) ? c0[dir * H2 + u0 + tid] : 0.0f;
  __syncthreads();

  const int lr  = tid >> 2;        // gate row 0..63
  const int seg = tid & 3;         // 128-wide K segment
  float* hbase = hbuf + dir * (2 * H2);

  for (int s = 0; s < T_LEN; ++s) {
    const int t = dir ? (T_LEN - 1 - s) : s;

    const float* hp = hbase + (s & 1) * H2;
    for (int i = tid; i < H2; i += 256) hprev[i] = hp[i];
    __syncthreads();

    float sum = 0.0f;
    const float* wrow = Wl + lr * H2 + seg * 128;
    const float* hseg = hprev + seg * 128;
    #pragma unroll 8
    for (int k = 0; k < 128; ++k) sum = fmaf(wrow[k], hseg[k], sum);
    partial[tid] = sum;
    __syncthreads();

    if (tid < ROWS_WG) {
      int q = tid >> 4, j = tid & 15;
      int r = q * H2 + u0 + j;
      gate[tid] = partial[tid * 4] + partial[tid * 4 + 1] +
                  partial[tid * 4 + 2] + partial[tid * 4 + 3] +
                  G[(size_t)t * RGATE + r];
    }
    __syncthreads();

    if (tid < UNITS_WG) {
      float gi = gate[tid], gf = gate[16 + tid], gg = gate[32 + tid], go = gate[48 + tid];
      float si = 1.0f / (1.0f + expf(-gi));
      float sf = 1.0f / (1.0f + expf(-gf));
      float so = 1.0f / (1.0f + expf(-go));
      creg = sf * creg + si * tanhf(gg);
      float h = so * tanhf(creg);
      hbase[((s + 1) & 1) * H2 + u0 + tid] = h;
      hs[(size_t)t * H2 + u0 + tid] = h;
    }
    grid_sync(cnt, phs);
  }
}

// ---------------------------------------------------------------------------
// Kernel 3: output projection  feats[t,n] = [hf;hb][t] . Wout[n] + bout[n]
// ---------------------------------------------------------------------------
__global__ void __launch_bounds__(128)
k_feats(const float* __restrict__ hsf, const float* __restrict__ hsb,
        const float* __restrict__ Wout, const float* __restrict__ bout,
        float* __restrict__ feats)
{
  const int t = blockIdx.x, tid = threadIdx.x;
  __shared__ float red[NTAGS][128];
  float p[NTAGS] = {0, 0, 0, 0, 0};
  for (int k = tid; k < 2 * H2; k += 128) {
    float x = (k < H2) ? hsf[(size_t)t * H2 + k] : hsb[(size_t)t * H2 + (k - H2)];
    #pragma unroll
    for (int n = 0; n < NTAGS; ++n) p[n] = fmaf(x, Wout[n * (2 * H2) + k], p[n]);
  }
  #pragma unroll
  for (int n = 0; n < NTAGS; ++n) red[n][tid] = p[n];
  __syncthreads();
  for (int off = 64; off > 0; off >>= 1) {
    if (tid < off)
      #pragma unroll
      for (int n = 0; n < NTAGS; ++n) red[n][tid] += red[n][tid + off];
    __syncthreads();
  }
  if (tid < NTAGS) feats[t * NTAGS + tid] = red[tid][0] + bout[tid];
}

// ---------------------------------------------------------------------------
// Kernel 4: Viterbi scan + backtrace, single workgroup.
// feats staged to LDS in 1024-step chunks; backpointers packed 3b/tag into
// ushort[T] (8 KB LDS) so backtrace never touches global memory.
// d_out[0] = path_score ; d_out[1..T] = best path tags (as float)
// ---------------------------------------------------------------------------
#define VCHUNK 1024
__global__ void __launch_bounds__(256)
k_viterbi(const float* __restrict__ feats, const float* __restrict__ trans,
          float* __restrict__ out)
{
  __shared__ float fch[VCHUNK][NTAGS];     // 20 KB
  __shared__ unsigned short bpk[T_LEN];    // 8 KB
  __shared__ float trs[NTAGS * NTAGS];
  __shared__ float fv[NTAGS], nfv[NTAGS];
  __shared__ unsigned char bps[NTAGS];

  const int tid = threadIdx.x;
  if (tid < NTAGS * NTAGS) trs[tid] = trans[tid];
  if (tid < NTAGS) fv[tid] = (tid == START_TAG) ? 0.0f : NEG_INF;
  __syncthreads();

  for (int c = 0; c < T_LEN / VCHUNK; ++c) {
    for (int i = tid; i < VCHUNK * NTAGS; i += 256)
      ((float*)fch)[i] = feats[c * VCHUNK * NTAGS + i];
    __syncthreads();

    for (int tt = 0; tt < VCHUNK; ++tt) {
      if (tid < NTAGS) {
        float best = -3.4e38f; int arg = 0;
        #pragma unroll
        for (int p = 0; p < NTAGS; ++p) {
          float sc = fv[p] + trs[tid * NTAGS + p];
          if (sc > best) { best = sc; arg = p; }
        }
        nfv[tid] = best + fch[tt][tid];
        bps[tid] = (unsigned char)arg;
      }
      __syncthreads();
      if (tid == 0)
        bpk[c * VCHUNK + tt] = (unsigned short)(bps[0] | (bps[1] << 3) | (bps[2] << 6) |
                                                (bps[3] << 9) | (bps[4] << 12));
      if (tid < NTAGS) fv[tid] = nfv[tid];
      __syncthreads();
    }
  }

  if (tid == 0) {
    float best = -3.4e38f; int arg = 0;
    #pragma unroll
    for (int n = 0; n < NTAGS; ++n) {
      float sc = fv[n] + trs[STOP_TAG * NTAGS + n];
      if (sc > best) { best = sc; arg = n; }
    }
    out[0] = best;
    int tag = arg;
    for (int t = T_LEN - 1; t >= 0; --t) {
      out[1 + t] = (float)tag;
      tag = (bpk[t] >> (3 * tag)) & 7;
    }
  }
}

// ---------------------------------------------------------------------------
// host-side orchestration
// ---------------------------------------------------------------------------
extern "C" void kernel_launch(void* const* d_in, const int* in_sizes, int n_in,
                              void* d_out, int out_size, void* d_ws, size_t ws_size,
                              hipStream_t stream)
{
  (void)in_sizes; (void)n_in; (void)out_size; (void)ws_size;
  const int*   sentence = (const int*)  d_in[0];
  const float* h0       = (const float*)d_in[1];
  const float* c0       = (const float*)d_in[2];
  const float* embed    = (const float*)d_in[3];
  const float* Wih_f    = (const float*)d_in[4];
  const float* Whh_f    = (const float*)d_in[5];
  const float* bih_f    = (const float*)d_in[6];
  const float* bhh_f    = (const float*)d_in[7];
  const float* Wih_b    = (const float*)d_in[8];
  const float* Whh_b    = (const float*)d_in[9];
  const float* bih_b    = (const float*)d_in[10];
  const float* bhh_b    = (const float*)d_in[11];
  const float* Wout     = (const float*)d_in[12];
  const float* bout     = (const float*)d_in[13];
  const float* trans    = (const float*)d_in[14];
  float* out = (float*)d_out;

  // bump allocator over workspace (256B aligned)
  uintptr_t p = (uintptr_t)d_ws;
  auto alloc = [&](size_t bytes) -> void* {
    void* r = (void*)p;
    p += (bytes + 255) & ~(size_t)255;
    return r;
  };
  unsigned short* Xbf  = (unsigned short*)alloc((size_t)T_LEN * EMB * 2);
  unsigned short* Wfb  = (unsigned short*)alloc((size_t)RGATE * EMB * 2);
  unsigned short* Wbb  = (unsigned short*)alloc((size_t)RGATE * EMB * 2);
  float* bsf   = (float*)alloc(RGATE * 4);
  float* bsb   = (float*)alloc(RGATE * 4);
  float* Gf    = (float*)alloc((size_t)T_LEN * RGATE * 4);
  float* Gb    = (float*)alloc((size_t)T_LEN * RGATE * 4);
  float* hsf   = (float*)alloc((size_t)T_LEN * H2 * 4);
  float* hsb   = (float*)alloc((size_t)T_LEN * H2 * 4);
  float* hbuf  = (float*)alloc(2 * 2 * H2 * 4);
  unsigned* sync = (unsigned*)alloc(256);
  float* feats = (float*)alloc((size_t)T_LEN * NTAGS * 4);

  // 0) preprocessing
  k_gather_bf16<<<T_LEN, 256, 0, stream>>>(sentence, embed, Xbf);
  k_cvt_bf16<<<(RGATE * EMB + 255) / 256, 256, 0, stream>>>(Wih_f, Wfb, RGATE * EMB);
  k_cvt_bf16<<<(RGATE * EMB + 255) / 256, 256, 0, stream>>>(Wih_b, Wbb, RGATE * EMB);
  k_bias_sum<<<(RGATE + 255) / 256, 256, 0, stream>>>(bih_f, bhh_f, bsf, RGATE);
  k_bias_sum<<<(RGATE + 255) / 256, 256, 0, stream>>>(bih_b, bhh_b, bsb, RGATE);
  k_init<<<4, 256, 0, stream>>>(h0, hbuf, sync);

  // 1) input GEMMs (both directions), 64x128 block tiles, async-LDS staged
  k_gemm_wmma<<<dim3(T_LEN / BM, RGATE / BN, 2), 256, 0, stream>>>(
      Xbf, Wfb, Wbb, bsf, bsb, Gf, Gb);

  // 2) persistent bidirectional recurrence (dynamic LDS: 128KB weights + staging)
  const size_t recLds = (size_t)(ROWS_WG * H2 + H2 + 256 + ROWS_WG) * sizeof(float);
  k_lstm_rec<<<2 * NWG_DIR, 256, recLds, stream>>>(
      Whh_f, Whh_b, c0, Gf, Gb, hbuf, hsf, hsb, sync);

  // 3) tag projection
  k_feats<<<T_LEN, 128, 0, stream>>>(hsf, hsb, Wout, bout, feats);

  // 4) Viterbi decode + backtrace
  k_viterbi<<<1, 256, 0, stream>>>(feats, trans, out);
}
